// MLoss_41240275976791
// MI455X (gfx1250) — compile-verified, matrix operand order checked by reading, not run on hardware
//
#include <hip/hip_runtime.h>
#include <hip/hip_bf16.h>

typedef __attribute__((ext_vector_type(16))) _Float16 v16h;
typedef __attribute__((ext_vector_type(4)))  _Float16 v4h;
typedef __attribute__((ext_vector_type(8)))  float    v8f;

#define DIM 512
#define CT 64            // classes per block
#define NT 256           // x-rows per chunk (8 waves x 2 frags x 16)
#define LDS_STRIDE 528   // 512 + 16 halves padding (keeps 32B align, spreads banks)
#define S_SCALE 64.0f

__device__ __forceinline__ float wave_sum32(float v) {
#pragma unroll
  for (int m = 16; m >= 1; m >>= 1) v += __shfl_xor(v, m, 32);
  return v;
}

__device__ __forceinline__ v8f wmma_f16(v16h a, v16h b, v8f c) {
  return __builtin_amdgcn_wmma_f32_16x16x32_f16(false, a, false, b, (short)0, c,
                                                false, false);
}

// Kernel 1: L2-normalize x rows, split into f16 hi/lo, store 1/||x||, zero sumexp.
__global__ __launch_bounds__(256) void norm_split_x(
    const float* __restrict__ x, _Float16* __restrict__ xh, _Float16* __restrict__ xl,
    float* __restrict__ x_inv, float* __restrict__ sumexp, int N) {
  int gid = blockIdx.x * 256 + threadIdx.x;
  if (gid < N) sumexp[gid] = 0.0f;               // re-zero accumulators every launch
  int i = blockIdx.x * 8 + (threadIdx.x >> 5);
  int lane = threadIdx.x & 31;
  if (i >= N) return;
  const float4* xp = (const float4*)(x + (size_t)i * DIM + lane * 16);
  float4 v[4];
  float s = 0.f;
#pragma unroll
  for (int k = 0; k < 4; ++k) {
    v[k] = xp[k];
    s += v[k].x * v[k].x + v[k].y * v[k].y + v[k].z * v[k].z + v[k].w * v[k].w;
  }
  s = wave_sum32(s);
  float inv = 1.0f / fmaxf(sqrtf(s), 1e-12f);
  if (lane == 0) x_inv[i] = inv;
  size_t base = (size_t)i * DIM + lane * 16;
#pragma unroll
  for (int k = 0; k < 4; ++k) {
    float f[4] = {v[k].x * inv, v[k].y * inv, v[k].z * inv, v[k].w * inv};
    v4h hh, ll;
#pragma unroll
    for (int j = 0; j < 4; ++j) {
      _Float16 h = (_Float16)f[j];
      hh[j] = h;
      ll[j] = (_Float16)(f[j] - (float)h);    // residual; f16 denorms preserved by WMMA
    }
    *(v4h*)(xh + base + 4 * k) = hh;
    *(v4h*)(xl + base + 4 * k) = ll;
  }
}

// Kernel 2: 1/||W_row|| for all C rows. One wave per row.
__global__ __launch_bounds__(256) void norm_w_rows(
    const float* __restrict__ W, float* __restrict__ w_inv, int C) {
  int i = blockIdx.x * 8 + (threadIdx.x >> 5);
  int lane = threadIdx.x & 31;
  if (i >= C) return;
  const float4* wp = (const float4*)(W + (size_t)i * DIM + lane * 16);
  float s = 0.f;
#pragma unroll
  for (int k = 0; k < 4; ++k) {
    float4 t = wp[k];
    s += t.x * t.x + t.y * t.y + t.z * t.z + t.w * t.w;
  }
  s = wave_sum32(s);
  if (lane == 0) w_inv[i] = 1.0f / fmaxf(sqrtf(s), 1e-12f);
}

// Kernel 3: true-class cosine in full fp32 (one wave per sample).
__global__ __launch_bounds__(256) void true_logits(
    const float* __restrict__ x, const float* __restrict__ W,
    const int* __restrict__ labels, const float* __restrict__ x_inv,
    const float* __restrict__ w_inv, float* __restrict__ tlog, int N) {
  int i = blockIdx.x * 8 + (threadIdx.x >> 5);
  int lane = threadIdx.x & 31;
  if (i >= N) return;
  int l = labels[i];
  const float4* xp = (const float4*)(x + (size_t)i * DIM);
  const float4* wp = (const float4*)(W + (size_t)l * DIM);
  float s = 0.f;
#pragma unroll
  for (int k = 0; k < 4; ++k) {
    float4 a = xp[lane + 32 * k];
    float4 b = wp[lane + 32 * k];
    s += a.x * b.x + a.y * b.y + a.z * b.z + a.w * b.w;
  }
  s = wave_sum32(s);
  if (lane == 0) tlog[i] = s * x_inv[i] * w_inv[l];
}

// Kernel 4: fused GEMM (split-f16 WMMA) + exp-row-sum accumulation.
// Block = 64 classes (full K in LDS as hi/lo f16), loops all N rows in 256-chunks.
// ks loop is kept ROLLED so B fragments are re-read from LDS each iteration
// (prevents the compiler from hoisting them across nb and spilling to scratch).
__global__ __launch_bounds__(256) void arc_gemm(
    const _Float16* __restrict__ xh, const _Float16* __restrict__ xl,
    const float* __restrict__ W, const float* __restrict__ w_inv,
    float* __restrict__ sumexp, int N, int C) {
  __shared__ _Float16 whs[CT * LDS_STRIDE];
  __shared__ _Float16 wls[CT * LDS_STRIDE];

  const int tid = threadIdx.x;
  const int c0 = blockIdx.x * CT;

  // Stage W tile: coalesced float4 loads, scale by 1/||row||, split hi/lo to LDS.
#pragma unroll 2
  for (int it = 0; it < 32; ++it) {
    int f = it * 256 + tid;        // flat float4 index over 64 x 128 grid
    int r = f >> 7;                // local class row 0..63
    int c4 = f & 127;              // float4 column 0..127
    int cls = c0 + r;
    float4 v = make_float4(0.f, 0.f, 0.f, 0.f);
    float inv = 0.f;
    if (cls < C) {
      v = ((const float4*)(W + (size_t)cls * DIM))[c4];
      inv = w_inv[cls];
    }
    float fv[4] = {v.x * inv, v.y * inv, v.z * inv, v.w * inv};
    v4h hh, ll;
#pragma unroll
    for (int j = 0; j < 4; ++j) {
      _Float16 h = (_Float16)fv[j];
      hh[j] = h;
      ll[j] = (_Float16)(fv[j] - (float)h);
    }
    *(v4h*)&whs[r * LDS_STRIDE + c4 * 4] = hh;
    *(v4h*)&wls[r * LDS_STRIDE + c4 * 4] = ll;
  }
  __syncthreads();

  const int wave = tid >> 5;
  const int lane = tid & 31;
  const int lrow = lane & 15;            // A: row M; B: column N; within 16-tile
  const int koff = (lane >> 4) * 16;     // K half selected by lane group

#pragma unroll 1
  for (int nb = 0; nb < N; nb += NT) {
    int row0 = nb + wave * 32 + lrow;    // first 16-row fragment for this wave
    const _Float16* xrh0 = xh + (size_t)row0 * DIM + koff;
    const _Float16* xrl0 = xl + (size_t)row0 * DIM + koff;
    const _Float16* xrh1 = xrh0 + (size_t)16 * DIM;
    const _Float16* xrl1 = xrl0 + (size_t)16 * DIM;

    v8f acc[2][4];
#pragma unroll
    for (int g = 0; g < 2; ++g)
#pragma unroll
      for (int t = 0; t < 4; ++t) acc[g][t] = v8f{};

#pragma unroll 1
    for (int ks = 0; ks < DIM; ks += 32) {
      v16h ah0 = *(const v16h*)(xrh0 + ks);
      v16h al0 = *(const v16h*)(xrl0 + ks);
      v16h ah1 = *(const v16h*)(xrh1 + ks);
      v16h al1 = *(const v16h*)(xrl1 + ks);
#pragma unroll
      for (int t = 0; t < 4; ++t) {
        int boff = (t * 16 + lrow) * LDS_STRIDE + ks + koff;
        v16h bh = *(const v16h*)&whs[boff];
        v16h bl = *(const v16h*)&wls[boff];
        acc[0][t] = wmma_f16(ah0, bh, acc[0][t]);
        acc[0][t] = wmma_f16(ah0, bl, acc[0][t]);
        acc[0][t] = wmma_f16(al0, bh, acc[0][t]);
        acc[1][t] = wmma_f16(ah1, bh, acc[1][t]);
        acc[1][t] = wmma_f16(ah1, bl, acc[1][t]);
        acc[1][t] = wmma_f16(al1, bh, acc[1][t]);
      }
    }

    // Epilogue: exp(S*wf), mask padded classes, reduce across the 16 columns,
    // one atomicAdd per (row, block).
#pragma unroll
    for (int g = 0; g < 2; ++g) {
#pragma unroll
      for (int r = 0; r < 8; ++r) {
        float e = 0.f;
#pragma unroll
        for (int t = 0; t < 4; ++t) {
          int cls = c0 + t * 16 + lrow;
          if (cls < C) e += __expf(S_SCALE * acc[g][t][r]);
        }
#pragma unroll
        for (int m = 8; m >= 1; m >>= 1) e += __shfl_xor(e, m, 32);
        if (lrow == 0)
          atomicAdd(&sumexp[nb + wave * 32 + g * 16 + r + 8 * (lane >> 4)], e);
      }
    }
  }
}

// Kernel 5: arcface margin + log-sum-exp assembly + mean reduction.
__global__ __launch_bounds__(256) void arc_finalize(
    const float* __restrict__ tlog, const float* __restrict__ sumexp,
    float* __restrict__ out, int N) {
  __shared__ float red[256];
  const float cM = 0.8775825618903728f;   // cos(0.5)
  const float sM = 0.4794255386042030f;   // sin(0.5)
  float acc = 0.f;
  for (int i = threadIdx.x; i < N; i += 256) {
    float t = tlog[i];
    float tc = fminf(fmaxf(t, -1.f + 1e-7f), 1.f - 1e-7f);
    float num = S_SCALE * (tc * cM - sqrtf(fmaxf(1.f - tc * tc, 0.f)) * sM);
    float den = fmaxf(__expf(num) + (sumexp[i] - __expf(S_SCALE * t)), 1e-10f);
    acc += num - __logf(den);
  }
  red[threadIdx.x] = acc;
  __syncthreads();
  for (int s2 = 128; s2 > 0; s2 >>= 1) {
    if (threadIdx.x < s2) red[threadIdx.x] += red[threadIdx.x + s2];
    __syncthreads();
  }
  if (threadIdx.x == 0) out[0] = -red[0] / (float)N;
}

extern "C" void kernel_launch(void* const* d_in, const int* in_sizes, int n_in,
                              void* d_out, int out_size, void* d_ws, size_t ws_size,
                              hipStream_t stream) {
  const float* x = (const float*)d_in[0];
  const float* W = (const float*)d_in[1];
  const int* labels = (const int*)d_in[2];   // jnp int64 -> int32 (x64 disabled)
  int N = in_sizes[0] / DIM;                 // 2048
  int C = in_sizes[1] / DIM;                 // 50000

  // Workspace layout (~4.3 MB)
  char* ws = (char*)d_ws;
  _Float16* xh = (_Float16*)ws;
  _Float16* xl = (_Float16*)(ws + (size_t)N * DIM * 2);
  float* x_inv = (float*)(ws + (size_t)N * DIM * 4);
  float* tlog  = x_inv + N;
  float* sume  = tlog + N;
  float* w_inv = sume + N;                   // C floats

  norm_split_x<<<(N + 7) / 8, 256, 0, stream>>>(x, xh, xl, x_inv, sume, N);
  norm_w_rows<<<(C + 7) / 8, 256, 0, stream>>>(W, w_inv, C);
  true_logits<<<(N + 7) / 8, 256, 0, stream>>>(x, W, labels, x_inv, w_inv, tlog, N);
  arc_gemm<<<(C + CT - 1) / CT, 256, 0, stream>>>(xh, xl, W, w_inv, sume, N, C);
  arc_finalize<<<1, 256, 0, stream>>>(tlog, sume, (float*)d_out, N);
}
